// VarianceAdaptorDurationOnly_48369921687653
// MI455X (gfx1250) — matrix operand order repaired and checked by reading, not compiled, not run on hardware
//
#include <hip/hip_runtime.h>
#include <hip/hip_bf16.h>

// Problem constants (specialized for this shape set)
#define BB   32
#define SS   512
#define HH   512
#define FF   512      // FILT
#define KW   3
#define LL   2048
#define KT   (HH*KW)  // 1536 = reduction length for both convs
#define EPSL 1e-5f
#define XPAD 40       // padded bf16 row stride for x slab (80B: 16B-aligned, bank-conflict-free)

typedef __bf16 bf16_t;
typedef __attribute__((ext_vector_type(16))) __bf16 v16bf;
typedef __attribute__((ext_vector_type(2)))  __bf16 v2bf;
typedef __attribute__((ext_vector_type(8)))  float  v8f;

// Pack two floats into packed bf16x2 with a single v_cvt_pk_bf16_f32 when the
// builtin is available; otherwise fall back to scalar casts (still cvt-based).
__device__ __forceinline__ unsigned pack_bf16x2(float a, float b) {
#if __has_builtin(__builtin_amdgcn_cvt_pk_bf16_f32)
  union { v2bf v; unsigned u; } p;
  p.v = __builtin_amdgcn_cvt_pk_bf16_f32(a, b);
  return p.u;
#else
  union { bf16_t h[2]; unsigned u; } p;
  p.h[0] = (bf16_t)a;
  p.h[1] = (bf16_t)b;
  return p.u;
#endif
}

// Repack conv weights [F][H][3] f32 -> k-major bf16 [F][k*512+h] so that both
// the A-fragment loads and the activation slab loads are fully contiguous.
__global__ __launch_bounds__(256)
void prep_w_kernel(const float* __restrict__ w, bf16_t* __restrict__ wp) {
  int tid = blockIdx.x * 256 + threadIdx.x;       // 0 .. 512*512-1
  int f = tid >> 9, h = tid & 511;
  const float* src = w + (size_t)f * KT + h * 3;
  bf16_t* dst = wp + (size_t)f * KT + h;
  dst[0]    = (bf16_t)src[0];
  dst[512]  = (bf16_t)src[1];
  dst[1024] = (bf16_t)src[2];
}

// A fragment 16x32 bf16 (MxK), loaded straight from global prepacked weights.
// ISA layout: lanes 0-15 row M=lane, K=0..7 & 16..23; lanes 16-31 row M=lane-16,
// K=8..15 & 24..31  ->  per lane two 16B chunks at element offsets kb and kb+16.
__device__ __forceinline__ v16bf load_a_direct(const bf16_t* rowptr, int off_el, int kb) {
  const uint4* p = (const uint4*)(rowptr + off_el + kb);
  union { uint4 q; bf16_t h[8]; } lo, hi;
  lo.q = p[0];
  hi.q = p[2];                                    // +16 elements = +32B
  v16bf a;
#pragma unroll
  for (int e = 0; e < 8; ++e) { a[e] = lo.h[e]; a[e + 8] = hi.h[e]; }
  return a;
}

// B fragment 32x16 bf16 (KxN) from the LDS x slab ([s][kc], row stride XPAD).
// Lanes 0-15: K=0..7 & 16..23, lanes 16-31: K=8..15 & 24..31 (matches A fragment).
__device__ __forceinline__ v16bf load_b_frag(const bf16_t* xb, int nt, int lane) {
  int n  = nt * 16 + (lane & 15);
  int i0 = (lane >> 4);
  const uint4* p = (const uint4*)(xb + n * XPAD);
  union { uint4 q; bf16_t h[8]; } u0, u1;
  u0.q = p[i0]; u1.q = p[i0 + 2];
  v16bf b;
#pragma unroll
  for (int e = 0; e < 8; ++e) { b[e] = u0.h[e]; b[e + 8] = u1.h[e]; }
  return b;
}

// MODE 0: conv1 (f32 in)  -> ReLU -> LN -> write h (bf16, packed b128 stores)
// MODE 1: conv2 (bf16 in) -> ReLU -> LN -> dot(lin_w)+lin_b -> masked log_dur
//         conv2 stages its x slab with GLOBAL_LOAD_ASYNC_TO_LDS (ASYNCcnt).
template <int MODE>
__global__ __launch_bounds__(256)
void conv_ln_kernel(const float* __restrict__ xf, const bf16_t* __restrict__ xh,
                    const bf16_t* __restrict__ wp, const float* __restrict__ cbias,
                    const float* __restrict__ gam, const float* __restrict__ bet,
                    const float* __restrict__ lin_w, const float* __restrict__ lin_b,
                    const unsigned char* __restrict__ src_mask,
                    bf16_t* __restrict__ hout, float* __restrict__ ld_out)
{
  __shared__ __align__(16) bf16_t xbuf[2][32 * XPAD];   // 2 x 2.5KB double buffer
  __shared__ float red1[256], red2[256], mu_s[32], rs_s[32];

  const int t    = threadIdx.x;
  const int lane = t & 31;
  const int wv   = t >> 5;              // wave owns filters [wv*64, wv*64+64)
  const int b    = blockIdx.y;
  const int s0   = blockIdx.x * 32;     // 32-position tile
  const int l15  = lane & 15;
  const int kb   = (lane >> 4) * 8;     // A-frag chunk selector / D-layout half

  // staging coords: each thread owns row srow, 8B segment q (fixed all loop)
  const int srow = t >> 3;              // 0..31
  const int q    = t & 7;               // 0..7  (q*4 elements into the 32-wide slab)

  // per-lane A row pointers (4 M-subtiles)
  const bf16_t* arow[4];
#pragma unroll
  for (int mt = 0; mt < 4; ++mt)
    arow[mt] = wp + (size_t)(wv * 64 + mt * 16 + l15) * KT;

  const v8f z = {0.f, 0.f, 0.f, 0.f, 0.f, 0.f, 0.f, 0.f};
  v8f acc[4][2];
#pragma unroll
  for (int mt = 0; mt < 4; ++mt) { acc[mt][0] = z; acc[mt][1] = z; }

  // chunk c: k = c>>4, h0 = (c&15)*32 ; reduction index off = k*512+h0
  auto stage = [&](int c, int buf) {
    int k  = c >> 4;
    int h0 = (c & 15) << 5;
    int sp = s0 + srow + k - 1;
    bool valid = (sp >= 0) && (sp < SS);
    if constexpr (MODE == 0) {
      uint2 o; o.x = 0u; o.y = 0u;
      if (valid) {
        float4 v = ((const float4*)(xf + ((size_t)b * SS + sp) * HH + h0))[q];
        o.x = pack_bf16x2(v.x, v.y);
        o.y = pack_bf16x2(v.z, v.w);
      }
      ((uint2*)(xbuf[buf] + srow * XPAD))[q] = o;
    } else {
      // bf16 pass-through: use the CDNA5 async global->LDS copy (ASYNCcnt).
      unsigned lds_b =
          (unsigned)(uintptr_t)(xbuf[buf] + srow * XPAD) + (unsigned)(q * 8);
      if (valid) {
        unsigned long long ga =
            (unsigned long long)(uintptr_t)(xh + ((size_t)b * SS + sp) * FF + h0 + q * 4);
        asm volatile("global_load_async_to_lds_b64 %0, %1, off"
                     :: "v"(lds_b), "v"(ga)
                     : "memory");
      } else {
        uint2 zz; zz.x = 0u; zz.y = 0u;
        ((uint2*)(xbuf[buf] + srow * XPAD))[q] = zz;
      }
    }
  };

  stage(0, 0);
  if constexpr (MODE == 1)
    asm volatile("s_wait_asynccnt 0x0" ::: "memory");
  __syncthreads();

  for (int c = 0; c < 48; ++c) {
    const int buf = c & 1;
    if (c + 1 < 48) stage(c + 1, buf ^ 1);      // prefetch next slab
    const int off = ((c >> 4) << 9) + ((c & 15) << 5);
    v16bf bf0 = load_b_frag(xbuf[buf], 0, lane);
    v16bf bf1 = load_b_frag(xbuf[buf], 1, lane);
#pragma unroll
    for (int mt = 0; mt < 4; ++mt) {
      v16bf af = load_a_direct(arow[mt], off, kb);
      acc[mt][0] = __builtin_amdgcn_wmma_f32_16x16x32_bf16(
          false, af, false, bf0, (short)0, acc[mt][0], false, false);
      acc[mt][1] = __builtin_amdgcn_wmma_f32_16x16x32_bf16(
          false, af, false, bf1, (short)0, acc[mt][1], false, false);
    }
    if constexpr (MODE == 1)
      asm volatile("s_wait_asynccnt 0x0" ::: "memory");
    __syncthreads();
  }

  // ---- epilogue: bias + ReLU (in registers, float4 param loads)
#pragma unroll
  for (int mt = 0; mt < 4; ++mt) {
    int fbase = wv * 64 + mt * 16 + kb;
    float cb[8];
    *(float4*)&cb[0] = *(const float4*)(cbias + fbase);
    *(float4*)&cb[4] = *(const float4*)(cbias + fbase + 4);
#pragma unroll
    for (int r = 0; r < 8; ++r)
#pragma unroll
      for (int nt = 0; nt < 2; ++nt) {
        float v = acc[mt][nt][r] + cb[r];
        acc[mt][nt][r] = v > 0.f ? v : 0.f;
      }
  }

  // ---- LayerNorm stats over the 512-filter axis, per position n
#pragma unroll
  for (int nt = 0; nt < 2; ++nt) {
    float s1 = 0.f, s2 = 0.f;
#pragma unroll
    for (int mt = 0; mt < 4; ++mt)
#pragma unroll
      for (int r = 0; r < 8; ++r) { float v = acc[mt][nt][r]; s1 += v; s2 += v * v; }
    s1 += __shfl_xor(s1, 16, 32);      // combine lane pair sharing same n
    s2 += __shfl_xor(s2, 16, 32);
    if (lane < 16) { red1[wv * 32 + nt * 16 + lane] = s1;
                     red2[wv * 32 + nt * 16 + lane] = s2; }
  }
  __syncthreads();
  if (t < 32) {
    float s1 = 0.f, s2 = 0.f;
    for (int w = 0; w < 8; ++w) { s1 += red1[w * 32 + t]; s2 += red2[w * 32 + t]; }
    float mu  = s1 * (1.f / FF);
    float var = s2 * (1.f / FF) - mu * mu;
    mu_s[t] = mu;
    rs_s[t] = rsqrtf(var + EPSL);
  }
  __syncthreads();

  if constexpr (MODE == 0) {
    // normalize and write bf16 h with packed 16B stores (8 consecutive f/lane)
#pragma unroll
    for (int mt = 0; mt < 4; ++mt) {
      int fbase = wv * 64 + mt * 16 + kb;
      float gg[8], bb[8];
      *(float4*)&gg[0] = *(const float4*)(gam + fbase);
      *(float4*)&gg[4] = *(const float4*)(gam + fbase + 4);
      *(float4*)&bb[0] = *(const float4*)(bet + fbase);
      *(float4*)&bb[4] = *(const float4*)(bet + fbase + 4);
#pragma unroll
      for (int nt = 0; nt < 2; ++nt) {
        int n = nt * 16 + l15;
        float mu = mu_s[n], rs = rs_s[n];
        float v[8];
#pragma unroll
        for (int r = 0; r < 8; ++r)
          v[r] = (acc[mt][nt][r] - mu) * rs * gg[r] + bb[r];
        uint4 o;
        o.x = pack_bf16x2(v[0], v[1]); o.y = pack_bf16x2(v[2], v[3]);
        o.z = pack_bf16x2(v[4], v[5]); o.w = pack_bf16x2(v[6], v[7]);
        *(uint4*)(hout + ((size_t)b * SS + s0 + n) * FF + fbase) = o;
      }
    }
  } else {
    // normalize, fuse 512->1 linear, masked log_dur write
    float d0 = 0.f, d1 = 0.f;
#pragma unroll
    for (int mt = 0; mt < 4; ++mt) {
      int fbase = wv * 64 + mt * 16 + kb;
      float gg[8], bb[8], lw[8];
      *(float4*)&gg[0] = *(const float4*)(gam + fbase);
      *(float4*)&gg[4] = *(const float4*)(gam + fbase + 4);
      *(float4*)&bb[0] = *(const float4*)(bet + fbase);
      *(float4*)&bb[4] = *(const float4*)(bet + fbase + 4);
      *(float4*)&lw[0] = *(const float4*)(lin_w + fbase);
      *(float4*)&lw[4] = *(const float4*)(lin_w + fbase + 4);
      int n0 = l15, n1 = 16 + l15;
#pragma unroll
      for (int r = 0; r < 8; ++r) {
        d0 += ((acc[mt][0][r] - mu_s[n0]) * rs_s[n0] * gg[r] + bb[r]) * lw[r];
        d1 += ((acc[mt][1][r] - mu_s[n1]) * rs_s[n1] * gg[r] + bb[r]) * lw[r];
      }
    }
    d0 += __shfl_xor(d0, 16, 32);
    d1 += __shfl_xor(d1, 16, 32);
    if (lane < 16) { red1[wv * 32 + lane] = d0; red1[wv * 32 + 16 + lane] = d1; }
    __syncthreads();
    if (t < 32) {
      float s = 0.f;
      for (int w = 0; w < 8; ++w) s += red1[w * 32 + t];
      s += lin_b[0];
      int sg = b * SS + s0 + t;
      ld_out[sg] = src_mask[sg] ? 0.f : s;
    }
  }
}

// Length regulator: cumsum(dur) scan in LDS, upper-bound search per frame,
// then pure-bandwidth float2 gather-copy of H=512 floats per frame.
__global__ __launch_bounds__(256)
void length_regulator_kernel(const float* __restrict__ x, const int* __restrict__ dur,
                             const unsigned char* __restrict__ mel_mask,
                             float* __restrict__ out_exp, float* __restrict__ out_mask)
{
  __shared__ int sa[SS], sb[SS];
  __shared__ int idx_s[64];
  const int t  = threadIdx.x;
  const int b  = blockIdx.y;
  const int t0 = blockIdx.x * 64;

#pragma unroll
  for (int j = 0; j < 2; ++j) { int i = t + 256 * j; sa[i] = dur[b * SS + i]; }
  __syncthreads();
  int* src = sa; int* dst = sb;
  for (int off = 1; off < SS; off <<= 1) {           // Hillis-Steele inclusive scan
#pragma unroll
    for (int j = 0; j < 2; ++j) {
      int i = t + 256 * j;
      int v = src[i];
      if (i >= off) v += src[i - off];
      dst[i] = v;
    }
    __syncthreads();
    int* tmp = src; src = dst; dst = tmp;
  }
  const int total = src[SS - 1];

  if (t < 64) {
    int frame = t0 + t;
    int lo = 0, hi = SS;                              // searchsorted(side='right')
    while (lo < hi) { int mid = (lo + hi) >> 1; if (src[mid] <= frame) lo = mid + 1; else hi = mid; }
    int id = lo < SS ? lo : SS - 1;
    idx_s[t] = (frame < total) ? id : -1;
    out_mask[b * LL + frame] = mel_mask[b * LL + frame] ? 1.f : 0.f;
  }
  __syncthreads();

  for (int fi = 0; fi < 64; ++fi) {
    int id = idx_s[fi];
    float2 v; v.x = 0.f; v.y = 0.f;
    if (id >= 0) v = ((const float2*)(x + ((size_t)b * SS + id) * HH))[t];
    ((float2*)(out_exp + ((size_t)b * LL + t0 + fi) * HH))[t] = v;
  }
}

extern "C" void kernel_launch(void* const* d_in, const int* in_sizes, int n_in,
                              void* d_out, int out_size, void* d_ws, size_t ws_size,
                              hipStream_t stream)
{
  const float*         x        = (const float*)d_in[0];
  const unsigned char* src_mask = (const unsigned char*)d_in[1];
  const unsigned char* mel_mask = (const unsigned char*)d_in[2];
  // d_in[3] = max_len scalar (2048, hardcoded), d_in[4..5] pitch/energy unused
  const int*   dur = (const int*)d_in[6];
  const float* c1w = (const float*)d_in[7];
  const float* c1b = (const float*)d_in[8];
  const float* g1  = (const float*)d_in[9];
  const float* b1  = (const float*)d_in[10];
  const float* c2w = (const float*)d_in[11];
  const float* c2b = (const float*)d_in[12];
  const float* g2  = (const float*)d_in[13];
  const float* b2  = (const float*)d_in[14];
  const float* lw  = (const float*)d_in[15];
  const float* lb  = (const float*)d_in[16];

  float* out      = (float*)d_out;
  float* out_exp  = out;                               // [B, L, H]
  float* out_ld   = out + (size_t)BB * LL * HH;        // [B, S]
  float* out_mask = out_ld + (size_t)BB * SS;          // [B, L]

  // bf16 scratch: h [B,S,FF] + two prepacked weight matrices [FF,KT].
  const size_t n_h = (size_t)BB * SS * FF;             // 8,388,608
  const size_t n_w = (size_t)FF * KT;                  //   786,432
  const size_t need = (n_h + 2 * n_w) * sizeof(bf16_t);
  bf16_t* base = (ws_size >= need) ? (bf16_t*)d_ws : (bf16_t*)out_exp; // out_exp is
  bf16_t* hbuf = base;                                 // rewritten by kernel3 later
  bf16_t* w1p  = base + n_h;
  bf16_t* w2p  = w1p + n_w;

  prep_w_kernel<<<(FF * HH) / 256, 256, 0, stream>>>(c1w, w1p);
  prep_w_kernel<<<(FF * HH) / 256, 256, 0, stream>>>(c2w, w2p);

  dim3 cgrid(SS / 32, BB);
  conv_ln_kernel<0><<<cgrid, 256, 0, stream>>>(
      x, nullptr, w1p, c1b, g1, b1, nullptr, nullptr, nullptr, hbuf, nullptr);
  conv_ln_kernel<1><<<cgrid, 256, 0, stream>>>(
      nullptr, hbuf, w2p, c2b, g2, b2, lw, lb, src_mask, nullptr, out_ld);

  dim3 lgrid(LL / 64, BB);
  length_regulator_kernel<<<lgrid, 256, 0, stream>>>(x, dur, mel_mask, out_exp, out_mask);
}
